// Sae_40785009443286
// MI455X (gfx1250) — compile-verified, matrix operand order checked by reading, not run on hardware
//
#include <hip/hip_runtime.h>
#include <hip/hip_bf16.h>

// ---------------------------------------------------------------------------
// TopK SAE forward for MI455X (gfx1250, wave32, WMMA).
//   pre = (x - b_dec) @ W_enc^T + b_enc ; feat = relu(pre)
//   (top_acts, top_idx) = topk(feat, 64) ; sae_out = top @ W_dec[idx] + b_dec
//   fvu, l2 from error. Outputs concatenated f32 in d_out.
// GEMM path: split-bf16 (hi+lo) with V_WMMA_F32_16X16X32_BF16 (3 products per
// k-step, ~fp32 accuracy). 64x64 wave tiles (4x4 WMMA accs) -> 48 WMMAs per
// 32 b128 loads per k-step. Falls back to single-bf16 if workspace is small.
// ---------------------------------------------------------------------------

#define D_IN        1024
#define NUM_LATENTS 32768
#define TOPK        64
#define BATCH       4096

typedef __bf16 bf16_t;
typedef bf16_t v16bf __attribute__((ext_vector_type(16)));
typedef float  v8f   __attribute__((ext_vector_type(8)));

union Frag16 { uint4 u[2]; v16bf v; };

__device__ __forceinline__ unsigned short f2bf_rne(float f) {
    unsigned u = __float_as_uint(f);
    unsigned r = u + 0x7FFFu + ((u >> 16) & 1u);
    return (unsigned short)(r >> 16);
}
__device__ __forceinline__ float bf2f(unsigned short h) {
    return __uint_as_float(((unsigned)h) << 16);
}

// ---- prep: xc = x - b_dec -> bf16 hi (+lo) ---------------------------------
__global__ __launch_bounds__(256) void sae_prep_x(
    const float* __restrict__ x, const float* __restrict__ b_dec,
    unsigned short* __restrict__ xh, unsigned short* __restrict__ xl,
    int use_split)
{
    int i = blockIdx.x * 256 + threadIdx.x;          // exactly BATCH*D_IN
    float v = x[i] - b_dec[i & (D_IN - 1)];
    unsigned short h = f2bf_rne(v);
    xh[i] = h;
    if (use_split) xl[i] = f2bf_rne(v - bf2f(h));
}

// ---- prep: W_enc -> bf16 hi (+lo) ------------------------------------------
__global__ __launch_bounds__(256) void sae_prep_w(
    const float* __restrict__ w,
    unsigned short* __restrict__ wh, unsigned short* __restrict__ wl,
    int use_split)
{
    int i = blockIdx.x * 256 + threadIdx.x;          // exactly NUM_LATENTS*D_IN
    float v = w[i];
    unsigned short h = f2bf_rne(v);
    wh[i] = h;
    if (use_split) wl[i] = f2bf_rne(v - bf2f(h));
}

// ---- fragment load: 16-bit A/B layout, K-contiguous row-major source -------
// lane&15 = M (or N) within the tile; lane>=16 selects K halves 8..15 /
// 24..31 per the CDNA5 16-bit A-matrix VGPR layout. Two 16B loads per frag.
__device__ __forceinline__ v16bf load_frag(const unsigned short* __restrict__ base,
                                           int row, int k0, int sel)
{
    const unsigned short* p = base + (size_t)row * D_IN + k0 + sel * 8;
    Frag16 f;
    f.u[0] = *reinterpret_cast<const uint4*>(p);        // K k0+sel*8 .. +7
    f.u[1] = *reinterpret_cast<const uint4*>(p + 16);   // K k0+16+sel*8 .. +7
    return f.v;
}

// ---- encoder GEMM + bias + relu -> feature_acts ----------------------------
// block = 256 threads = 8 waves in 4(M) x 2(N); wave tile 64x64 (4x4 accs);
// workgroup tile 256(M) x 128(N); grid = (256 N-blocks, 16 M-blocks).
__global__ __launch_bounds__(256) void sae_gemm(
    const unsigned short* __restrict__ Wh, const unsigned short* __restrict__ Wl,
    const unsigned short* __restrict__ Xh, const unsigned short* __restrict__ Xl,
    const float* __restrict__ b_enc, float* __restrict__ feat, int use_split)
{
    const int lane = threadIdx.x & 31;
    const int wid  = threadIdx.x >> 5;
    const int wm   = wid & 3;            // 4 waves along M
    const int wn   = wid >> 2;           // 2 waves along N
    const int m_base = blockIdx.y * 256 + wm * 64;
    const int n_base = blockIdx.x * 128 + wn * 64;
    const int sel = (lane >> 4) & 1;
    const int lr  = lane & 15;

    v8f acc[4][4];
    const v8f zero = {0.f,0.f,0.f,0.f,0.f,0.f,0.f,0.f};
#pragma unroll
    for (int m = 0; m < 4; ++m)
#pragma unroll
        for (int n = 0; n < 4; ++n) acc[m][n] = zero;

    for (int k0 = 0; k0 < D_IN; k0 += 32) {
        v16bf ah[4], bh[4];
#pragma unroll
        for (int m = 0; m < 4; ++m)
            ah[m] = load_frag(Xh, m_base + m * 16 + lr, k0, sel);
#pragma unroll
        for (int n = 0; n < 4; ++n)
            bh[n] = load_frag(Wh, n_base + n * 16 + lr, k0, sel);

        if (use_split) {
            v16bf al[4], bl[4];
#pragma unroll
            for (int m = 0; m < 4; ++m)
                al[m] = load_frag(Xl, m_base + m * 16 + lr, k0, sel);
#pragma unroll
            for (int n = 0; n < 4; ++n)
                bl[n] = load_frag(Wl, n_base + n * 16 + lr, k0, sel);
#pragma unroll
            for (int m = 0; m < 4; ++m)
#pragma unroll
                for (int n = 0; n < 4; ++n) {
                    acc[m][n] = __builtin_amdgcn_wmma_f32_16x16x32_bf16(
                        false, ah[m], false, bh[n], (short)0, acc[m][n], false, false);
                    acc[m][n] = __builtin_amdgcn_wmma_f32_16x16x32_bf16(
                        false, ah[m], false, bl[n], (short)0, acc[m][n], false, false);
                    acc[m][n] = __builtin_amdgcn_wmma_f32_16x16x32_bf16(
                        false, al[m], false, bh[n], (short)0, acc[m][n], false, false);
                }
        } else {
#pragma unroll
            for (int m = 0; m < 4; ++m)
#pragma unroll
                for (int n = 0; n < 4; ++n)
                    acc[m][n] = __builtin_amdgcn_wmma_f32_16x16x32_bf16(
                        false, ah[m], false, bh[n], (short)0, acc[m][n], false, false);
        }
    }

    // epilogue: C layout -> VGPR r: M = r + (lane>=16 ? 8 : 0), N = lane&15
#pragma unroll
    for (int n = 0; n < 4; ++n) {
        const int col  = n_base + n * 16 + lr;
        const float bias = b_enc[col];
#pragma unroll
        for (int m = 0; m < 4; ++m) {
            const int row0 = m_base + m * 16 + sel * 8;
#pragma unroll
            for (int r = 0; r < 8; ++r) {
                float v = acc[m][n][r] + bias;
                v = v > 0.f ? v : 0.f;
                feat[(size_t)(row0 + r) * NUM_LATENTS + col] = v;
            }
        }
    }
}

// ---- exact per-row top-64 via LDS-staged 4-round radix select --------------
// All values >= 0 after relu, so float bits compare like unsigned ints.
__global__ __launch_bounds__(256) void sae_topk(
    const float* __restrict__ feat,
    float* __restrict__ top_acts, float* __restrict__ top_idx)
{
    __shared__ float    vals[NUM_LATENTS];    // 128 KB (of 320 KB LDS per WGP)
    __shared__ unsigned hist[256];
    __shared__ unsigned s_prefix, s_remaining, s_cgt, s_ceq;

    const int tid = threadIdx.x;
    const int row = blockIdx.x;

    const float4* src = reinterpret_cast<const float4*>(feat + (size_t)row * NUM_LATENTS);
    float4* dst = reinterpret_cast<float4*>(vals);
    for (int i = tid; i < NUM_LATENTS / 4; i += 256) dst[i] = src[i];
    if (tid == 0) { s_prefix = 0u; s_remaining = TOPK; s_cgt = 0u; s_ceq = 0u; }
    __syncthreads();

    for (int round = 0; round < 4; ++round) {
        const int shift = 24 - 8 * round;
        hist[tid] = 0u;
        __syncthreads();
        const unsigned prefix = s_prefix;
        const unsigned mask = (round == 0) ? 0u : (0xFFFFFFFFu << (shift + 8));
        for (int i = tid; i < NUM_LATENTS; i += 256) {
            unsigned u = __float_as_uint(vals[i]);
            if ((u & mask) == prefix) atomicAdd(&hist[(u >> shift) & 255u], 1u);
        }
        __syncthreads();
        if (tid == 0) {
            unsigned cum = 0, rem = s_remaining;
            for (int d = 255; d >= 0; --d) {
                unsigned h = hist[d];
                if (cum + h >= rem) {
                    s_remaining = rem - cum;
                    s_prefix = prefix | ((unsigned)d << shift);
                    break;
                }
                cum += h;
            }
        }
        __syncthreads();
    }

    const unsigned T   = s_prefix;       // exact bits of the K-th largest value
    const unsigned rem = s_remaining;    // how many ties at T to keep
    const unsigned ngt = TOPK - rem;
    for (int i = tid; i < NUM_LATENTS; i += 256) {
        unsigned u = __float_as_uint(vals[i]);
        int slot = -1;
        if (u > T) {
            slot = (int)atomicAdd(&s_cgt, 1u);
        } else if (u == T) {
            unsigned p = atomicAdd(&s_ceq, 1u);
            if (p < rem) slot = (int)(ngt + p);
        }
        if (slot >= 0) {
            top_acts[(size_t)row * TOPK + slot] = vals[i];
            top_idx [(size_t)row * TOPK + slot] = (float)i;
        }
    }
}

// ---- sparse decode + error accumulation ------------------------------------
__global__ __launch_bounds__(256) void sae_decode(
    const float* __restrict__ x, const float* __restrict__ W_dec,
    const float* __restrict__ b_dec,
    const float* __restrict__ top_acts, const float* __restrict__ top_idx,
    float* __restrict__ sae_out, float* __restrict__ err_total)
{
    __shared__ float s_act[TOPK];
    __shared__ int   s_idx[TOPK];
    __shared__ float s_red[256];

    const int tid = threadIdx.x;
    const int row = blockIdx.x;
    if (tid < TOPK) {
        s_act[tid] = top_acts[(size_t)row * TOPK + tid];
        s_idx[tid] = (int)top_idx[(size_t)row * TOPK + tid];
    }
    __syncthreads();

    // prefetch the first gathered decoder rows into cache (global_prefetch_b8)
    __builtin_prefetch(W_dec + (size_t)s_idx[0] * D_IN + tid, 0, 1);
    __builtin_prefetch(W_dec + (size_t)s_idx[1] * D_IN + tid, 0, 1);

    float a0 = b_dec[tid], a1 = b_dec[tid + 256], a2 = b_dec[tid + 512], a3 = b_dec[tid + 768];
#pragma unroll 4
    for (int k = 0; k < TOPK; ++k) {
        if (k + 8 < TOPK)   // hide gather latency: prefetch row 8 steps ahead
            __builtin_prefetch(W_dec + (size_t)s_idx[k + 8] * D_IN + tid, 0, 1);
        const float a = s_act[k];
        const float* wr = W_dec + (size_t)s_idx[k] * D_IN;
        a0 += a * wr[tid];
        a1 += a * wr[tid + 256];
        a2 += a * wr[tid + 512];
        a3 += a * wr[tid + 768];
    }
    const float* xr = x + (size_t)row * D_IN;
    float* orow = sae_out + (size_t)row * D_IN;
    float e0 = a0 - xr[tid], e1 = a1 - xr[tid + 256];
    float e2 = a2 - xr[tid + 512], e3 = a3 - xr[tid + 768];
    orow[tid] = a0; orow[tid + 256] = a1; orow[tid + 512] = a2; orow[tid + 768] = a3;

    s_red[tid] = e0 * e0 + e1 * e1 + e2 * e2 + e3 * e3;
    __syncthreads();
    for (int s = 128; s > 0; s >>= 1) {
        if (tid < s) s_red[tid] += s_red[tid + s];
        __syncthreads();
    }
    if (tid == 0) atomicAdd(err_total, s_red[0]);
}

// ---- x column sums + sum of squares (for total_variance) -------------------
__global__ __launch_bounds__(256) void sae_stats(
    const float* __restrict__ x, float* __restrict__ colsum, float* __restrict__ sumsq)
{
    __shared__ float s_red[256];
    const int tid = threadIdx.x;
    const int r0  = blockIdx.x * 64;
    float c0 = 0.f, c1 = 0.f, c2 = 0.f, c3 = 0.f, sq = 0.f;
    for (int r = 0; r < 64; ++r) {
        const float* xr = x + (size_t)(r0 + r) * D_IN;
        float v0 = xr[tid], v1 = xr[tid + 256], v2 = xr[tid + 512], v3 = xr[tid + 768];
        c0 += v0; c1 += v1; c2 += v2; c3 += v3;
        sq += v0 * v0 + v1 * v1 + v2 * v2 + v3 * v3;
    }
    atomicAdd(&colsum[tid], c0);
    atomicAdd(&colsum[tid + 256], c1);
    atomicAdd(&colsum[tid + 512], c2);
    atomicAdd(&colsum[tid + 768], c3);
    s_red[tid] = sq;
    __syncthreads();
    for (int s = 128; s > 0; s >>= 1) {
        if (tid < s) s_red[tid] += s_red[tid + s];
        __syncthreads();
    }
    if (tid == 0) atomicAdd(sumsq, s_red[0]);
}

__global__ void sae_zero(float* p, int n)
{
    int i = blockIdx.x * 256 + threadIdx.x;
    if (i < n) p[i] = 0.f;
}

__global__ void sae_finalize(const float* __restrict__ colsum,
                             const float* __restrict__ sumsq,
                             const float* __restrict__ err,
                             float* __restrict__ tail)
{
    if (threadIdx.x == 0 && blockIdx.x == 0) {
        float s2 = 0.f;
        for (int c = 0; c < D_IN; ++c) { float m = colsum[c]; s2 += m * m; }
        float tv = *sumsq - s2 / (float)BATCH;   // sum((x - mean_col)^2)
        float e  = *err;
        tail[0] = e / tv;            // fvu
        tail[1] = 0.f;               // auxk
        tail[2] = 0.f;               // multi_topk
        tail[3] = 0.f;               // l1
        tail[4] = e / (float)BATCH;  // l2_loss
    }
}

extern "C" void kernel_launch(void* const* d_in, const int* in_sizes, int n_in,
                              void* d_out, int out_size, void* d_ws, size_t ws_size,
                              hipStream_t stream)
{
    (void)in_sizes; (void)n_in; (void)out_size;
    const float* x     = (const float*)d_in[0];
    const float* W_enc = (const float*)d_in[1];
    const float* b_enc = (const float*)d_in[2];
    const float* W_dec = (const float*)d_in[3];
    const float* b_dec = (const float*)d_in[4];

    float* out      = (float*)d_out;
    float* sae_out  = out;                                   // 4096*1024
    float* feat     = out + (size_t)BATCH * D_IN;            // 4096*32768
    float* top_acts = feat + (size_t)BATCH * NUM_LATENTS;    // 4096*64
    float* top_idx  = top_acts + (size_t)BATCH * TOPK;       // 4096*64 (as f32)
    float* tail     = top_idx + (size_t)BATCH * TOPK;        // 5 scalars

    // workspace layout (bytes): hi-precision path fits in ~75.5 MB;
    // split (hi+lo) path needs ~151 MB.
    unsigned char* ws = (unsigned char*)d_ws;
    const size_t OFF_WH    = 0;                                   // 64 MB
    const size_t OFF_XH    = OFF_WH + (size_t)NUM_LATENTS * D_IN * 2;   // +8 MB
    const size_t OFF_STATS = OFF_XH + (size_t)BATCH * D_IN * 2;         // 1026 f32
    const size_t OFF_WL    = ((OFF_STATS + 1026 * 4 + 15) / 16) * 16;   // 64 MB
    const size_t OFF_XL    = OFF_WL + (size_t)NUM_LATENTS * D_IN * 2;   // 8 MB
    const size_t NEED_SPLIT = OFF_XL + (size_t)BATCH * D_IN * 2;

    unsigned short* Wh = (unsigned short*)(ws + OFF_WH);
    unsigned short* Xh = (unsigned short*)(ws + OFF_XH);
    float*          st = (float*)(ws + OFF_STATS);   // colsum[1024], sumsq, err
    unsigned short* Wl = (unsigned short*)(ws + OFF_WL);
    unsigned short* Xl = (unsigned short*)(ws + OFF_XL);
    const int use_split = (ws_size >= NEED_SPLIT) ? 1 : 0;

    sae_zero<<<(1026 + 255) / 256, 256, 0, stream>>>(st, 1026);
    sae_prep_x<<<(BATCH * D_IN) / 256, 256, 0, stream>>>(x, b_dec, Xh, Xl, use_split);
    sae_prep_w<<<(NUM_LATENTS * D_IN) / 256, 256, 0, stream>>>(W_enc, Wh, Wl, use_split);
    sae_stats<<<BATCH / 64, 256, 0, stream>>>(x, st, st + 1024);

    dim3 ggrid(NUM_LATENTS / 128, BATCH / 256);
    sae_gemm<<<ggrid, 256, 0, stream>>>(Wh, Wl, Xh, Xl, b_enc, feat, use_split);

    sae_topk<<<BATCH, 256, 0, stream>>>(feat, top_acts, top_idx);
    sae_decode<<<BATCH, 256, 0, stream>>>(x, W_dec, b_dec, top_acts, top_idx,
                                          sae_out, st + 1025);
    sae_finalize<<<1, 32, 0, stream>>>(st, st + 1024, st + 1025, tail);
}
